// ResidualGroup_50732153700398
// MI455X (gfx1250) — matrix operand order repaired and checked by reading, not compiled, not run on hardware
//
#include <hip/hip_runtime.h>
#include <math.h>

// ---------------------------------------------------------------------------
// MI455X (gfx1250) implementation.
// Heavy GEMMs (im2col conv, patch correlation) use v_wmma_f32_16x16x32_bf16
// with a 64x64 block tile: B staged in LDS via the Tensor Data Mover
// (tensor_load_to_lds + s_wait_tensorcnt, double buffered), A streamed from
// global. Sparse top-3 attention is an atomic overlap-add fold (bandwidth
// bound), never a dense einsum. wave32 everywhere.
// ---------------------------------------------------------------------------

typedef __bf16 bf16_t;
typedef bf16_t v16bf __attribute__((ext_vector_type(16)));
typedef float  v8f   __attribute__((ext_vector_type(8)));
typedef unsigned int u32x4 __attribute__((ext_vector_type(4)));
typedef int i32x4 __attribute__((ext_vector_type(4)));
typedef int i32x8 __attribute__((ext_vector_type(8)));

#define CEILDIV(a,b) (((a)+(b)-1)/(b))

#if __has_builtin(__builtin_amdgcn_tensor_load_to_lds)
#define HAVE_TDM 1
#else
#define HAVE_TDM 0
#endif

__device__ __forceinline__ unsigned short f2bf(float f) {
  unsigned int u = __float_as_uint(f);
  u += 0x7FFFu + ((u >> 16) & 1u);          // round-to-nearest-even
  return (unsigned short)(u >> 16);
}

#if HAVE_TDM
// Issue a TDM load of a [rows<=64] x [32 elem] bf16 tile (row stride = K
// elements) from global into LDS at byte offset lds_off. 2-D descriptor,
// zero-fills rows beyond tensor_dim1 (N edge).
__device__ __forceinline__ void tdm_load_b_tile(unsigned lds_off, long long gbyte,
                                                int rows_left, int K)
{
  u32x4 g0;
  g0[0] = 1u;                                              // count=1, user D#
  g0[1] = lds_off;                                         // LDS byte address
  g0[2] = (unsigned)(gbyte & 0xFFFFFFFFll);                // global_addr[31:0]
  g0[3] = (unsigned)((gbyte >> 32) & 0x01FFFFFFll) | (2u << 30);  // addr hi | type=2
  unsigned td0 = (unsigned)K;                              // X extent (>=32)
  unsigned td1 = (unsigned)(rows_left < 0 ? 0 : rows_left);// Y extent (zero-fill)
  unsigned long long st0 = (unsigned long long)K;          // row stride (elems)
  i32x8 g1;
  g1[0] = (int)(1u << 16);                                 // data_size=1 (2 bytes)
  g1[1] = (int)((td0 & 0xFFFFu) << 16);                    // tensor_dim0 lo
  g1[2] = (int)(((td0 >> 16) & 0xFFFFu) | ((td1 & 0xFFFFu) << 16));
  g1[3] = (int)(((td1 >> 16) & 0xFFFFu) | (32u << 16));    // tile_dim0 = 32
  g1[4] = (int)64u;                                        // tile_dim1=64, tile_dim2=0
  g1[5] = (int)(st0 & 0xFFFFFFFFull);
  g1[6] = (int)((st0 >> 32) & 0xFFFFull);                  // dim1 stride unused
  g1[7] = 0;
  i32x4 gz = {0, 0, 0, 0};
#if defined(__clang_major__) && (__clang_major__ >= 23)
  i32x8 gz8 = {0, 0, 0, 0, 0, 0, 0, 0};
  __builtin_amdgcn_tensor_load_to_lds(g0, g1, gz, gz, gz8, 0);
#else
  __builtin_amdgcn_tensor_load_to_lds(g0, g1, gz, gz, 0);
#endif
}
#endif

// ---------------------------------------------------------------------------
// WMMA GEMM:  out[M,N] = A[M,K](bf16) * Wt[N,K]^T(bf16) + bias[N]
// Block tile 64(M) x 64(N): 4 waves x (16x64 per wave, 4 accumulators).
// B tile (64x32 bf16 per K step) staged in LDS by TDM (double buffered),
// shared by all waves. K must be a multiple of 32.
// mode 0: out row-major [M,N].  mode 1: NCHW scatter, m=(b,hw), n=channel.
// ---------------------------------------------------------------------------
__global__ void __launch_bounds__(128)
gemm_bf16_wmma(const unsigned short* __restrict__ A,
               const unsigned short* __restrict__ Wt,
               const float* __restrict__ bias,
               float* __restrict__ out,
               int M, int N, int K, int mode, int HW, int relu)
{
  __shared__ unsigned short bsm[2][64 * 32];      // double-buffered B tile, 8 KB
  const int wave = threadIdx.x >> 5;
  const int lane = threadIdx.x & 31;
  const int m0 = blockIdx.x * 64 + wave * 16;
  const int n0 = blockIdx.y * 64;

  int arow = m0 + (lane & 15);
  if (arow > M - 1) arow = M - 1;
  const int ka = (lane >> 4) * 8;                 // A: K sub-offset per lane half
  const int kb = (lane >> 4) * 8;                 // B: K sub-offset per lane half
  const unsigned short* ap = A + (size_t)arow * K + ka;

  v8f zero = {0.f, 0.f, 0.f, 0.f, 0.f, 0.f, 0.f, 0.f};
  v8f acc[4];
#pragma unroll
  for (int t = 0; t < 4; ++t) acc[t] = zero;

#if HAVE_TDM
  if (wave == 0) {
    tdm_load_b_tile((unsigned)(size_t)&bsm[0][0],
                    (long long)(size_t)Wt + ((long long)n0 * K) * 2, N - n0, K);
  }
#endif

  for (int kk = 0; kk < K; kk += 32) {
    const int cur = (kk >> 5) & 1;
#if HAVE_TDM
    if (wave == 0) {
      if (kk + 32 < K) {
        tdm_load_b_tile((unsigned)(size_t)&bsm[cur ^ 1][0],
                        (long long)(size_t)Wt + ((long long)n0 * K + (kk + 32)) * 2,
                        N - n0, K);
        __builtin_amdgcn_s_wait_tensorcnt((short)1);   // current buffer retired
      } else {
        __builtin_amdgcn_s_wait_tensorcnt((short)0);
      }
    }
#else
    // cooperative staging with plain loads (fallback when TDM builtin absent)
    for (int idx = threadIdx.x; idx < 64 * 4; idx += 128) {
      int row = idx >> 2, seg = idx & 3;
      int rr = n0 + row; if (rr > N - 1) rr = N - 1;
      *(uint4*)&bsm[cur][row * 32 + seg * 8] =
          *(const uint4*)(Wt + (size_t)rr * K + kk + seg * 8);
    }
#endif
    __syncthreads();                               // B tile visible to all waves

    union { v16bf v; uint4 q[2]; } afrag;
    // A 16x32 bf16 layout: lane half -> K {ka..ka+7} and {ka+16..ka+23}
    afrag.q[0] = *(const uint4*)(ap + kk);
    afrag.q[1] = *(const uint4*)(ap + kk + 16);

#pragma unroll
    for (int t = 0; t < 4; ++t) {
      union { v16bf v; uint4 q[2]; } bfrag;
      // B 32x16 layout: lane half -> K {kb..kb+7} and {kb+16..kb+23} for one column N
      const unsigned short* bp = &bsm[cur][(t * 16 + (lane & 15)) * 32 + kb];
      bfrag.q[0] = *(const uint4*)(bp);
      bfrag.q[1] = *(const uint4*)(bp + 16);
      acc[t] = __builtin_amdgcn_wmma_f32_16x16x32_bf16(
          false, afrag.v, false, bfrag.v, (short)0, acc[t], false, false);
    }
    __syncthreads();                               // done reading before overwrite
  }

  const int mbase = m0 + ((lane >> 4) << 3);
#pragma unroll
  for (int t = 0; t < 4; ++t) {
    const int nc = n0 + t * 16 + (lane & 15);
#pragma unroll
    for (int r = 0; r < 8; ++r) {
      int mr = mbase + r;
      if (mr < M && nc < N) {
        float v = acc[t][r];
        if (bias) v += bias[nc];
        if (relu) v = fmaxf(v, 0.f);
        size_t oi;
        if (mode == 1) {
          int b = mr / HW, p = mr - b * HW;
          oi = ((size_t)b * N + nc) * HW + p;
        } else {
          oi = (size_t)mr * N + nc;
        }
        out[oi] = v;
      }
    }
  }
}

// ---------------------------------------------------------------------------
// im2col (channel-major patch order, matches torch unfold / OIHW weights)
// col[M,K], M = B*OH*OW, K = C*k*k, bf16 output.
// ---------------------------------------------------------------------------
__global__ void im2col_bf16(const float* __restrict__ in, unsigned short* __restrict__ col,
                            int Bn, int C, int H, int W, int k, int s, int p,
                            int OH, int OW)
{
  long long K = (long long)C * k * k;
  long long total = (long long)Bn * OH * OW * K;
  long long i = (long long)blockIdx.x * blockDim.x + threadIdx.x;
  if (i >= total) return;
  long long m = i / K; int kk = (int)(i - m * K);
  int b = (int)(m / (OH * OW)); int pp = (int)(m % (OH * OW));
  int oy = pp / OW, ox = pp % OW;
  int c = kk / (k * k); int r = kk % (k * k); int dy = r / k, dx = r % k;
  int yy = oy * s - p + dy, xx = ox * s - p + dx;
  float v = (yy >= 0 && yy < H && xx >= 0 && xx < W)
            ? in[(((size_t)b * C + c) * H + yy) * W + xx] : 0.f;
  col[i] = f2bf(v);
}

// weights [Nsrc,K] f32 -> [Npad,K] bf16 (zero padded rows)
__global__ void cast_pad_bf16(const float* __restrict__ w, unsigned short* __restrict__ o,
                              int Nsrc, int Npad, int K)
{
  long long i = (long long)blockIdx.x * blockDim.x + threadIdx.x;
  if (i >= (long long)Npad * K) return;
  int n = (int)(i / K);
  o[i] = (n < Nsrc) ? f2bf(w[i]) : (unsigned short)0;
}

__global__ void pad_bias(const float* b, float* o, int n, int np) {
  int t = threadIdx.x;
  if (t < np) o[t] = (t < n) ? b[t] : 0.f;
}

// ---------------------------------------------------------------------------
// Elementwise / small kernels
// ---------------------------------------------------------------------------
__global__ void add2(const float* a, const float* b, float* o, long long n) {
  long long i = (long long)blockIdx.x * blockDim.x + threadIdx.x;
  if (i < n) o[i] = a[i] + b[i];
}

__global__ void apply_ca(const float* src, const float* h, const float* sc,
                         float* dst, int C, int HW, long long n) {
  long long i = (long long)blockIdx.x * blockDim.x + threadIdx.x;
  if (i >= n) return;
  int bc = (int)(i / HW);
  dst[i] = src[i] + h[i] * sc[bc];
}

__global__ void chan_mean(const float* in, float* out, int HW) {
  __shared__ float red[256];
  const float* p = in + (size_t)blockIdx.x * HW;
  float s = 0.f;
  for (int i = threadIdx.x; i < HW; i += 256) s += p[i];
  red[threadIdx.x] = s; __syncthreads();
  for (int o = 128; o > 0; o >>= 1) {
    if (threadIdx.x < o) red[threadIdx.x] += red[threadIdx.x + o];
    __syncthreads();
  }
  if (threadIdx.x == 0) out[blockIdx.x] = red[0] / (float)HW;
}

// channel attention MLP: 64 -> 4 (relu) -> 64 (sigmoid). one block per batch.
__global__ void ca_mlp(const float* s, const float* wd, const float* bd,
                       const float* wu, const float* bu, float* scale) {
  __shared__ float sv[64], dv[4];
  int b = blockIdx.x, t = threadIdx.x;
  sv[t] = s[b * 64 + t];
  __syncthreads();
  if (t < 4) {
    float a = bd[t];
    for (int c = 0; c < 64; ++c) a += wd[t * 64 + c] * sv[c];
    dv[t] = fmaxf(a, 0.f);
  }
  __syncthreads();
  float a = bu[t];
  for (int j = 0; j < 4; ++j) a += wu[t * 4 + j] * dv[j];
  scale[b * 64 + t] = 1.f / (1.f + expf(-a));
}

// ConvTranspose2d 64->64 k=8 s=4 p=2: exactly 2x2 taps contribute per output.
__global__ void convT8s4(const float* __restrict__ in, const float* __restrict__ w,
                         const float* __restrict__ bias, float* __restrict__ out,
                         int B, int C, int h, int wi, int OH, int OW)
{
  long long tot = (long long)B * C * OH * OW;
  long long i = (long long)blockIdx.x * blockDim.x + threadIdx.x;
  if (i >= tot) return;
  int X = (int)(i % OW); long long t = i / OW;
  int Y = (int)(t % OH); t /= OH;
  int co = (int)(t % C); int b = (int)(t / C);
  float acc = bias[co];
  int py = (Y + 2) & 3, px = (X + 2) & 3;
  for (int ky = py; ky < 8; ky += 4) {
    int yq = Y + 2 - ky; if (yq < 0) continue; int yy = yq >> 2; if (yy >= h) continue;
    for (int kx = px; kx < 8; kx += 4) {
      int xq = X + 2 - kx; if (xq < 0) continue; int xx = xq >> 2; if (xx >= wi) continue;
      for (int ci = 0; ci < C; ++ci)
        acc += in[(((size_t)b * C + ci) * h + yy) * wi + xx]
             * w[(((size_t)ci * C + co) * 8 + ky) * 8 + kx];
    }
  }
  out[i] = acc;
}

// pixel_unshuffle d=4: ref_s16 [B,16,192,192] (first 4 ch valid) -> [B,64,48,48]
__global__ void pix_unshuffle(const float* __restrict__ in, float* __restrict__ out,
                              long long n)
{
  long long i = (long long)blockIdx.x * blockDim.x + threadIdx.x;
  if (i >= n) return;
  int ww = (int)(i % 48); long long t = i / 48;
  int hh = (int)(t % 48); t /= 48;
  int oc = (int)(t % 64); int b = (int)(t / 64);
  int c = oc >> 4; int r = oc & 15; int yy = r >> 2; int xx = r & 3;
  out[i] = in[(((size_t)b * 16 + c) * 192 + hh * 4 + yy) * 192 + ww * 4 + xx];
}

__global__ void concat2(const float* a, const float* b, float* o,
                        int C1, int C2, int HW, long long n)
{
  long long i = (long long)blockIdx.x * blockDim.x + threadIdx.x;
  if (i >= n) return;
  int p = (int)(i % HW); long long t = i / HW;
  int c = (int)(t % (C1 + C2)); int bb = (int)(t / (C1 + C2));
  o[i] = (c < C1) ? a[((size_t)bb * C1 + c) * HW + p]
                  : b[((size_t)bb * C2 + (c - C1)) * HW + p];
}

// unfold (VALID, channel-major) + L2 row normalization -> bf16 [B*L, K]
__global__ void unfold_norm_bf16(const float* __restrict__ in, unsigned short* __restrict__ out,
                                 int C, int H, int W, int k, int s, int OH, int OW)
{
  extern __shared__ float sh[];            // K floats
  __shared__ float red[128];
  int L = OH * OW;
  int bl = blockIdx.x;
  int b = bl / L, l = bl % L;
  int ly = (l / OW) * s, lx = (l % OW) * s;
  int K = C * k * k;
  float ss = 0.f;
  for (int kk = threadIdx.x; kk < K; kk += blockDim.x) {
    int c = kk / (k * k); int r = kk % (k * k); int dy = r / k, dx = r % k;
    float v = in[(((size_t)b * C + c) * H + ly + dy) * W + lx + dx];
    sh[kk] = v; ss += v * v;
  }
  red[threadIdx.x] = ss; __syncthreads();
  for (int o = 64; o > 0; o >>= 1) {
    if (threadIdx.x < o) red[threadIdx.x] += red[threadIdx.x + o];
    __syncthreads();
  }
  float inv = 1.f / fmaxf(sqrtf(red[0]), 1e-12f);
  for (int kk = threadIdx.x; kk < K; kk += blockDim.x)
    out[(size_t)bl * K + kk] = f2bf(sh[kk] * inv);
}

// top-3 of each correlation row (value = score weight, per reference)
__global__ void top3_kernel(const float* __restrict__ corr, int L,
                            int* __restrict__ idx, float* __restrict__ wv, int boff)
{
  int l = blockIdx.x * blockDim.x + threadIdx.x;
  if (l >= L) return;
  float v0 = -1e30f, v1 = -1e30f, v2 = -1e30f; int i0 = 0, i1 = 0, i2 = 0;
  const float* row = corr + (size_t)l * L;
  for (int m = 0; m < L; ++m) {
    float v = row[m];
    if (v > v0)      { v2 = v1; i2 = i1; v1 = v0; i1 = i0; v0 = v; i0 = m; }
    else if (v > v1) { v2 = v1; i2 = i1; v1 = v;  i1 = m; }
    else if (v > v2) { v2 = v;  i2 = m; }
  }
  size_t o = (size_t)(boff + l) * 3;
  idx[o] = i0; idx[o + 1] = i1; idx[o + 2] = i2;
  wv[o]  = v0; wv[o + 1]  = v1; wv[o + 2]  = v2;
}

// sparse attention + fold overlap-add: acc[l-patch] += w_j * src[m_j-patch]
__global__ void sparse_fold(const float* __restrict__ src, float* __restrict__ acc,
                            const int* __restrict__ idx, const float* __restrict__ wv,
                            int C, int H, int W, int k, int s, int OH, int OW)
{
  int L = OH * OW;
  int b = blockIdx.x / L, l = blockIdx.x % L;
  int ly = (l / OW) * s, lx = (l % OW) * s;
  int K = C * k * k;
  for (int j = 0; j < 3; ++j) {
    int   m = idx[(size_t)(b * L + l) * 3 + j];
    float w = wv[(size_t)(b * L + l) * 3 + j];
    int my = (m / OW) * s, mx = (m % OW) * s;
    for (int kk = threadIdx.x; kk < K; kk += blockDim.x) {
      int c = kk / (k * k); int r = kk % (k * k); int dy = r / k, dx = r % k;
      atomicAdd(&acc[(((size_t)b * C + c) * H + ly + dy) * W + lx + dx],
                w * src[(((size_t)b * C + c) * H + my + dy) * W + mx + dx]);
    }
  }
}

// analytic fold(ones) coverage counts
__global__ void cover_kernel(float* cov, int H, int W, int k, int s, int OH, int OW) {
  int i = blockIdx.x * blockDim.x + threadIdx.x;
  if (i >= H * W) return;
  int y = i / W, x = i % W;
  int ay = y - k + 1; int loy = (ay <= 0) ? 0 : (ay + s - 1) / s;
  int hiy = y / s; if (hiy > OH - 1) hiy = OH - 1;
  int ax = x - k + 1; int lox = (ax <= 0) ? 0 : (ax + s - 1) / s;
  int hix = x / s; if (hix > OW - 1) hix = OW - 1;
  cov[i] = (float)((hiy - loy + 1) * (hix - lox + 1));
}

__global__ void div_cover(float* t, const float* cov, int HW, long long n) {
  long long i = (long long)blockIdx.x * blockDim.x + threadIdx.x;
  if (i >= n) return;
  t[i] /= cov[i % HW];
}

// ---------------------------------------------------------------------------
// Host-side helpers
// ---------------------------------------------------------------------------
struct BlkP { const float *w1,*b1,*w2,*b2,*wd,*bd,*wu,*bu; };

static void conv_gemm(hipStream_t st, const float* in, int B, int C, int H, int W,
                      int k, int s, int p, const float* w, const float* bias,
                      int Cout, int Npad, float* out, int relu,
                      unsigned short* colbuf, unsigned short* wbuf, float* biasbuf)
{
  int OH = (H + 2 * p - k) / s + 1, OW = (W + 2 * p - k) / s + 1;
  int M = B * OH * OW, K = C * k * k;
  long long tot = (long long)M * K;
  im2col_bf16<<<(unsigned)CEILDIV(tot, 256), 256, 0, st>>>(in, colbuf, B, C, H, W, k, s, p, OH, OW);
  cast_pad_bf16<<<(unsigned)CEILDIV((long long)Npad * K, 256), 256, 0, st>>>(w, wbuf, Cout, Npad, K);
  const float* bptr = bias;
  if (Npad != Cout) { pad_bias<<<1, Npad, 0, st>>>(bias, biasbuf, Cout, Npad); bptr = biasbuf; }
  dim3 g(CEILDIV(M, 64), CEILDIV(Npad, 64));
  gemm_bf16_wmma<<<g, 128, 0, st>>>(colbuf, wbuf, bptr, out, M, Npad, K, 1, OH * OW, relu);
}

static void run_body(hipStream_t st, const float* in, int B, int H, int W, float* dst,
                     const BlkP* blk, const float* tw, const float* tb,
                     float* t_h1, float* t_h2, float* t_cur, float* meanb, float* scaleb,
                     unsigned short* colbuf, unsigned short* wbuf, float* biasbuf)
{
  const int C = 64, HW = H * W;
  const float* cur = in;
  long long n = (long long)B * C * HW;
  for (int i = 0; i < 4; ++i) {
    conv_gemm(st, cur,  B, C, H, W, 3, 1, 1, blk[i].w1, blk[i].b1, 64, 64, t_h1, 1, colbuf, wbuf, biasbuf);
    conv_gemm(st, t_h1, B, C, H, W, 3, 1, 1, blk[i].w2, blk[i].b2, 64, 64, t_h2, 0, colbuf, wbuf, biasbuf);
    chan_mean<<<B * 64, 256, 0, st>>>(t_h2, meanb, HW);
    ca_mlp<<<B, 64, 0, st>>>(meanb, blk[i].wd, blk[i].bd, blk[i].wu, blk[i].bu, scaleb);
    apply_ca<<<(unsigned)CEILDIV(n, 256), 256, 0, st>>>(cur, t_h2, scaleb, t_cur, 64, HW, n);
    cur = t_cur;
  }
  conv_gemm(st, cur, B, C, H, W, 3, 1, 1, tw, tb, 64, 64, dst, 0, colbuf, wbuf, biasbuf);
}

// ---------------------------------------------------------------------------
extern "C" void kernel_launch(void* const* d_in, const int* in_sizes, int n_in,
                              void* d_out, int out_size, void* d_ws, size_t ws_size,
                              hipStream_t stream)
{
  const int B = 2, C = 64, Hs = 48, Ws = 48, Hb = 192, Wb = 192;
  const long long SZ_SM  = (long long)B * C * Hs * Ws;      //   294912
  const long long SZ_BIG = (long long)B * C * Hb * Wb;      //  4718592
  const int OH4 = 45, L = OH4 * OH4;                        // 2025 patches
  const int Kp = C * 16;                                    // 1024 patch dim

  const float* x  = (const float*)d_in[0];
  const float* y  = (const float*)d_in[1];
  const float* yd = (const float*)d_in[2];
  // params flattened: body.blocks[0..3]{b1,b2,bd,bu,w1,w2,wd,wu}, body.tb, body.tw,
  // compress_b/w, down2_b/w, last2_b/w, last_b/w, scale_b/w, up_b/w
  BlkP blk[4];
  for (int i = 0; i < 4; ++i) {
    int p = 3 + i * 8;
    blk[i].b1 = (const float*)d_in[p + 0];
    blk[i].b2 = (const float*)d_in[p + 1];
    blk[i].bd = (const float*)d_in[p + 2];
    blk[i].bu = (const float*)d_in[p + 3];
    blk[i].w1 = (const float*)d_in[p + 4];
    blk[i].w2 = (const float*)d_in[p + 5];
    blk[i].wd = (const float*)d_in[p + 6];
    blk[i].wu = (const float*)d_in[p + 7];
  }
  const float* tb         = (const float*)d_in[35];
  const float* tw         = (const float*)d_in[36];
  const float* compress_b = (const float*)d_in[37];
  const float* compress_w = (const float*)d_in[38];
  const float* down2_b    = (const float*)d_in[39];
  const float* down2_w    = (const float*)d_in[40];
  const float* last2_b    = (const float*)d_in[41];
  const float* last2_w    = (const float*)d_in[42];
  const float* last_b     = (const float*)d_in[43];
  const float* last_w     = (const float*)d_in[44];
  const float* scale_b    = (const float*)d_in[45];
  const float* scale_w    = (const float*)d_in[46];
  const float* up_b       = (const float*)d_in[47];
  const float* up_w       = (const float*)d_in[48];

  // ---- workspace bump allocator ----
  char* ws = (char*)d_ws;
  size_t off = 0;
  auto alloc = [&](size_t bytes) -> char* {
    char* p = ws + off;
    off = (off + bytes + 255) & ~(size_t)255;
    return p;
  };
  float* t_h1    = (float*)alloc(SZ_BIG * 4);
  float* t_h2    = (float*)alloc(SZ_BIG * 4);
  float* t_cur   = (float*)alloc(SZ_BIG * 4);
  float* ref     = (float*)alloc(SZ_BIG * 4);
  float* res     = (float*)alloc(SZ_SM * 4);
  float* refl    = (float*)alloc(SZ_SM * 4);
  float* res_up  = (float*)alloc(SZ_BIG * 4);
  float* up_acc  = (float*)alloc(SZ_BIG * 4);
  float* cmp_cat = (float*)alloc(2 * SZ_BIG * 4);
  float* cmp     = (float*)alloc(SZ_BIG * 4);
  float* ref_s16 = (float*)alloc((size_t)B * 16 * Hb * Wb * 4);
  float* ref_ps  = (float*)alloc(SZ_SM * 4);
  float* out_acc = (float*)alloc(SZ_SM * 4);
  float* cat_sm  = (float*)alloc(2 * SZ_SM * 4);
  float* out1    = (float*)alloc(SZ_SM * 4);
  float* out_dn  = (float*)alloc(SZ_SM * 4);
  float* out2    = (float*)alloc(SZ_SM * 4);
  float* corrbuf = (float*)alloc((size_t)L * L * 4);
  float* cover16 = (float*)alloc((size_t)Hb * Wb * 4);
  float* cover4  = (float*)alloc((size_t)Hs * Ws * 4);
  float* meanb   = (float*)alloc(B * 64 * 4);
  float* scaleb  = (float*)alloc(B * 64 * 4);
  float* biasbuf = (float*)alloc(64 * 4);
  float* top3w   = (float*)alloc((size_t)B * L * 3 * 4);
  int*   top3i   = (int*)  alloc((size_t)B * L * 3 * 4);
  unsigned short* colbuf = (unsigned short*)alloc((size_t)(B * Hb * Wb) * 576 * 2);
  unsigned short* wbuf   = (unsigned short*)alloc((size_t)64 * 4096 * 2);
  unsigned short* lrn    = (unsigned short*)alloc((size_t)B * L * Kp * 2);
  unsigned short* rfn    = (unsigned short*)alloc((size_t)B * L * Kp * 2);

  float* o_main = (float*)d_out;
  float* o_ref1 = o_main + SZ_SM;
  float* o_refd = o_ref1 + SZ_BIG;

  // ---- shared residual body on x, y, y_down ----
  run_body(stream, x,  B, Hs, Ws, res,  blk, tw, tb, t_h1, t_h2, t_cur, meanb, scaleb, colbuf, wbuf, biasbuf);
  run_body(stream, y,  B, Hb, Wb, ref,  blk, tw, tb, t_h1, t_h2, t_cur, meanb, scaleb, colbuf, wbuf, biasbuf);
  run_body(stream, yd, B, Hs, Ws, refl, blk, tw, tb, t_h1, t_h2, t_cur, meanb, scaleb, colbuf, wbuf, biasbuf);

  // ref1 = ref + y ; ref_down = refl + y_down
  add2<<<(unsigned)CEILDIV(SZ_BIG, 256), 256, 0, stream>>>(ref,  y,  o_ref1, SZ_BIG);
  add2<<<(unsigned)CEILDIV(SZ_SM,  256), 256, 0, stream>>>(refl, yd, o_refd, SZ_SM);

  // res_up = ConvTranspose2d(res) k8 s4 p2 -> [B,64,192,192]
  convT8s4<<<(unsigned)CEILDIV(SZ_BIG, 256), 256, 0, stream>>>(res, up_w, up_b, res_up, B, C, Hs, Ws, Hb, Wb);

  // scale conv (64->4, padded to 16 lanes for WMMA) then pixel_unshuffle(4)
  conv_gemm(stream, ref, B, C, Hb, Wb, 3, 1, 1, scale_w, scale_b, 4, 16, ref_s16, 0, colbuf, wbuf, biasbuf);
  pix_unshuffle<<<(unsigned)CEILDIV(SZ_SM, 256), 256, 0, stream>>>(ref_s16, ref_ps, SZ_SM);

  // normalized patch matrices (bf16) for cosine correlation
  unfold_norm_bf16<<<B * L, 128, Kp * 4, stream>>>(res,  lrn, C, Hs, Ws, 4, 1, OH4, OH4);
  unfold_norm_bf16<<<B * L, 128, Kp * 4, stream>>>(refl, rfn, C, Hs, Ws, 4, 1, OH4, OH4);

  hipMemsetAsync(out_acc, 0, SZ_SM * 4, stream);
  hipMemsetAsync(up_acc,  0, SZ_BIG * 4, stream);

  // per-batch correlation GEMM (2025x2025x1024, WMMA) + top-3
  for (int b = 0; b < B; ++b) {
    dim3 g(CEILDIV(L, 64), CEILDIV(L, 64));
    gemm_bf16_wmma<<<g, 128, 0, stream>>>(lrn + (size_t)b * L * Kp,
                                          rfn + (size_t)b * L * Kp,
                                          nullptr, corrbuf, L, L, Kp, 0, 0, 0);
    top3_kernel<<<CEILDIV(L, 128), 128, 0, stream>>>(corrbuf, L, top3i, top3w, b * L);
  }

  // sparse attention folds (overlap-add with atomics)
  sparse_fold<<<B * L, 256, 0, stream>>>(ref_ps, out_acc, top3i, top3w, C, Hs, Ws, 4,  1, OH4, OH4);
  sparse_fold<<<B * L, 256, 0, stream>>>(ref,    up_acc,  top3i, top3w, C, Hb, Wb, 16, 4, OH4, OH4);

  // divide by fold(ones) coverage
  cover_kernel<<<CEILDIV(Hs * Ws, 256), 256, 0, stream>>>(cover4,  Hs, Ws, 4,  1, OH4, OH4);
  cover_kernel<<<CEILDIV(Hb * Wb, 256), 256, 0, stream>>>(cover16, Hb, Wb, 16, 4, OH4, OH4);
  div_cover<<<(unsigned)CEILDIV(SZ_SM,  256), 256, 0, stream>>>(out_acc, cover4,  Hs * Ws, SZ_SM);
  div_cover<<<(unsigned)CEILDIV(SZ_BIG, 256), 256, 0, stream>>>(up_acc,  cover16, Hb * Wb, SZ_BIG);

  // compress(concat(res_up, out_up)) 1x1 128->64, then down2 k8 s4 p2 -> 48x48
  concat2<<<(unsigned)CEILDIV(2 * SZ_BIG, 256), 256, 0, stream>>>(res_up, up_acc, cmp_cat, C, C, Hb * Wb, 2 * SZ_BIG);
  conv_gemm(stream, cmp_cat, B, 2 * C, Hb, Wb, 1, 1, 0, compress_w, compress_b, 64, 64, cmp, 0, colbuf, wbuf, biasbuf);
  conv_gemm(stream, cmp, B, C, Hb, Wb, 8, 4, 2, down2_w, down2_b, 64, 64, out_dn, 0, colbuf, wbuf, biasbuf);

  // last(concat(res, out)) 3x3 128->64
  concat2<<<(unsigned)CEILDIV(2 * SZ_SM, 256), 256, 0, stream>>>(res, out_acc, cat_sm, C, C, Hs * Ws, 2 * SZ_SM);
  conv_gemm(stream, cat_sm, B, 2 * C, Hs, Ws, 3, 1, 1, last_w, last_b, 64, 64, out1, 0, colbuf, wbuf, biasbuf);

  // last2(concat(out, out_up)) 3x3 128->64 + relu, then + x
  concat2<<<(unsigned)CEILDIV(2 * SZ_SM, 256), 256, 0, stream>>>(out1, out_dn, cat_sm, C, C, Hs * Ws, 2 * SZ_SM);
  conv_gemm(stream, cat_sm, B, 2 * C, Hs, Ws, 3, 1, 1, last2_w, last2_b, 64, 64, out2, 1, colbuf, wbuf, biasbuf);
  add2<<<(unsigned)CEILDIV(SZ_SM, 256), 256, 0, stream>>>(out2, x, o_main, SZ_SM);

  (void)in_sizes; (void)n_in; (void)out_size; (void)ws_size;
}
